// RayRoPEAttention_12008728560319
// MI455X (gfx1250) — matrix operand order repaired and checked by reading, not compiled, not run on hardware
//
#include <hip/hip_runtime.h>
#include <hip/hip_bf16.h>
#include <math.h>

// ---------------------------------------------------------------------------
// RayRoPE attention for MI455X (gfx1250, wave32, WMMA + TDM).
// Compute-bound (~70 GFLOP vs ~40MB of traffic at 23.3 TB/s) -> everything
// through v_wmma_f32_16x16x32_bf16 with f32 accumulation; V tiles staged to
// LDS by the Tensor Data Mover (tensor_load_to_lds / s_wait_tensorcnt).
// ---------------------------------------------------------------------------

typedef __bf16 bf16;
typedef __attribute__((ext_vector_type(16))) __bf16 v16bf;
typedef __attribute__((ext_vector_type(8)))  __bf16 v8bf;
typedef __attribute__((ext_vector_type(8)))  float  v8f;

static constexpr int Bc   = 2;
static constexpr int Fc   = 8;
static constexpr int Pc   = 256;
static constexpr int Sc   = 2048;   // F * P
static constexpr int Hc   = 16;
static constexpr int Dc   = 64;
static constexpr int DIMc = 1024;
static constexpr int HALFc = 32;

union V16U { v16bf v; v8bf h[2]; };

// ---- Tensor Data Mover support (toolchain-arity hedge) --------------------
#if defined(__has_builtin)
#  if __has_builtin(__builtin_amdgcn_tensor_load_to_lds)
#    define HAVE_TDM 1
#  endif
#endif
#ifndef HAVE_TDM
#  define HAVE_TDM 0
#endif
#if defined(__has_include)
#  if __has_include(<hip/amd_detail/amd_gfx1250_TDM.h>)
#    define TDM_SIX_ARGS 1   // amdgpu-toolchain / therock headers -> 6-arg builtin
#  else
#    define TDM_SIX_ARGS 0   // ROCm 7.2 -> 5-arg builtin
#  endif
#else
#  define TDM_SIX_ARGS 0
#endif

#if HAVE_TDM
typedef unsigned int u32x4 __attribute__((ext_vector_type(4)));
typedef int          i32x4 __attribute__((ext_vector_type(4)));
typedef int          i32x8 __attribute__((ext_vector_type(8)));

// DMA a dense 2-D tile (rows x rowElems, bf16, row-major, stride==rowElems)
// from global memory into LDS at byte offset lds_off.
__device__ inline void tdm_load_2d_bf16(unsigned lds_off, const void* gptr,
                                        unsigned rowElems, unsigned rows) {
  const unsigned long long ga = (unsigned long long)(uintptr_t)gptr;
  u32x4 g0;
  g0[0] = 1u;                                               // count=1 (valid user D#)
  g0[1] = lds_off;                                          // lds_addr
  g0[2] = (unsigned)(ga & 0xffffffffu);                     // global_addr[31:0]
  g0[3] = (unsigned)((ga >> 32) & 0x01ffffffu) | (2u << 30);// addr[56:32] | type=2
  i32x8 g1;
  g1[0] = (int)(1u << 16);          // workgroup_mask=0, data_size=1 (2 bytes)
  g1[1] = (int)(rowElems << 16);    // tensor_dim0 (bits 63:48)
  g1[2] = (int)(rows << 16);        // tensor_dim1 (bits 95:80)
  g1[3] = (int)(rowElems << 16);    // tile_dim0   (bits 127:112)
  g1[4] = (int)rows;                // tile_dim1   (bits 143:128)
  g1[5] = (int)rowElems;            // tensor_dim0_stride (bits 191:160)
  g1[6] = 0;
  g1[7] = 0;
  const i32x4 z4 = {0, 0, 0, 0};
#if TDM_SIX_ARGS
  const i32x8 z8 = {0, 0, 0, 0, 0, 0, 0, 0};
  __builtin_amdgcn_tensor_load_to_lds(g0, g1, z4, z4, z8, 0);
#else
  __builtin_amdgcn_tensor_load_to_lds(g0, g1, z4, z4, 0);
#endif
}
#endif  // HAVE_TDM

__device__ inline v8f wmma_bf16(v16bf a, v16bf b, v8f c) {
  // D = A(16x32 bf16) * B(32x16 bf16) + C(16x16 f32)
  return __builtin_amdgcn_wmma_f32_16x16x32_bf16(
      /*neg_a=*/false, a, /*neg_b=*/false, b,
      /*c_mod=*/(short)0, c, /*reuse_a=*/false, /*reuse_b=*/false);
}

// A-fragment (16x32, M x K): lane l holds row (l&15); lanes 0-15 carry
// K = {0..7, 16..23}, lanes 16-31 carry K = {8..15, 24..31}.
__device__ inline v16bf load_afrag(const bf16* base, int stride, int lane) {
  const int row = lane & 15;
  const int kb  = (lane < 16) ? 0 : 8;
  const bf16* p = base + row * stride + kb;
  V16U r;
  r.h[0] = *(const v8bf*)(p);
  r.h[1] = *(const v8bf*)(p + 16);
  return r.v;
}

// B-fragment (32x16, K x N) loaded from a K-contiguous (transposed) buffer:
// lane l holds col (l&15); lanes 0-15 carry K = {0..7, 16..23}, lanes 16-31 K = {8..15, 24..31}.
__device__ inline v16bf load_bfrag(const bf16* base, int stride, int lane) {
  const int col = lane & 15;
  const int kb  = (lane < 16) ? 0 : 8;
  const bf16* p = base + col * stride + kb;
  V16U r;
  r.h[0] = *(const v8bf*)(p);
  r.h[1] = *(const v8bf*)(p + 16);
  return r.v;
}

__device__ inline float halfmax(float v) {
#pragma unroll
  for (int m = 1; m <= 8; m <<= 1) v = fmaxf(v, __shfl_xor(v, m, 32));
  return v;
}
__device__ inline float halfsum(float v) {
#pragma unroll
  for (int m = 1; m <= 8; m <<= 1) v += __shfl_xor(v, m, 32);
  return v;
}

// ---------------------------------------------------------------------------
// Kernel 1: per-token RoPE cos/sin tables from camera rays.
// One block per (b,f); one thread per patch pixel.
// ---------------------------------------------------------------------------
__global__ __launch_bounds__(256) void rope_kernel(
    const float* __restrict__ w2cs, const float* __restrict__ Ks,
    float* __restrict__ CS, float* __restrict__ SN) {
  const int bf = blockIdx.x;          // 0..B*F-1
  const int b  = bf >> 3;
  const int f  = bf & 7;
  const float* M  = w2cs + bf * 16;
  const float* Kc = Ks   + bf * 9;

  float R[3][3], t[3];
#pragma unroll
  for (int i = 0; i < 3; ++i) {
#pragma unroll
    for (int j = 0; j < 3; ++j) R[i][j] = M[i * 4 + j];
    t[i] = M[i * 4 + 3];
  }
  float o[3];
#pragma unroll
  for (int i = 0; i < 3; ++i)
    o[i] = -(R[0][i] * t[0] + R[1][i] * t[1] + R[2][i] * t[2]);

  // analytic 3x3 inverse of intrinsics
  const float a00 = Kc[0], a01 = Kc[1], a02 = Kc[2];
  const float a10 = Kc[3], a11 = Kc[4], a12 = Kc[5];
  const float a20 = Kc[6], a21 = Kc[7], a22 = Kc[8];
  const float c00 = a11 * a22 - a12 * a21;
  const float c01 = -(a10 * a22 - a12 * a20);
  const float c02 = a10 * a21 - a11 * a20;
  const float id  = 1.0f / (a00 * c00 + a01 * c01 + a02 * c02);
  float Ki[3][3];
  Ki[0][0] = c00 * id; Ki[0][1] = -(a01 * a22 - a02 * a21) * id; Ki[0][2] = (a01 * a12 - a02 * a11) * id;
  Ki[1][0] = c01 * id; Ki[1][1] =  (a00 * a22 - a02 * a20) * id; Ki[1][2] = -(a00 * a12 - a02 * a10) * id;
  Ki[2][0] = c02 * id; Ki[2][1] = -(a00 * a21 - a01 * a20) * id; Ki[2][2] = (a00 * a11 - a01 * a10) * id;

  const int p = threadIdx.x;
  const float u = (float)((p & 15)) * 16.0f + 8.0f;
  const float v = (float)((p >> 4)) * 16.0f + 8.0f;
  float dc[3], dw[3];
#pragma unroll
  for (int i = 0; i < 3; ++i) dc[i] = Ki[i][0] * u + Ki[i][1] * v + Ki[i][2];
#pragma unroll
  for (int i = 0; i < 3; ++i)
    dw[i] = R[0][i] * dc[0] + R[1][i] * dc[1] + R[2][i] * dc[2];
  const float inv = rsqrtf(dw[0] * dw[0] + dw[1] * dw[1] + dw[2] * dw[2]);
  const float d0 = dw[0] * inv, d1 = dw[1] * inv, d2 = dw[2] * inv;
  const float m0 = o[1] * d2 - o[2] * d1;
  const float m1 = o[2] * d0 - o[0] * d2;
  const float m2 = o[0] * d1 - o[1] * d0;
  const float ray[6] = {d0, d1, d2, m0, m1, m2};

  const int s = f * Pc + p;
  float* cp = CS + ((size_t)b * Sc + s) * HALFc;
  float* sp = SN + ((size_t)b * Sc + s) * HALFc;
#pragma unroll 8
  for (int i = 0; i < HALFc; ++i) {
    const float fr = __powf(100.0f, -(float)i * (1.0f / (float)HALFc));
    const float a = ray[i % 6] * fr;
    float sv, cv;
    sincosf(a, &sv, &cv);
    cp[i] = cv;
    sp[i] = sv;
  }
}

// ---------------------------------------------------------------------------
// Kernel 2: tiled WMMA GEMM  C[2048x1024] = A[2048x1024] @ W[1024x1024] + bias
// Block tile 128x128, K step 32, 8 waves (4 M x 2 N), wave tile 32x64.
// OUT_MODE 0: rope + bf16 [B,H,S,D];  1: bf16 [B,H,S,D];  2: f32 [B,S,DIM].
// ---------------------------------------------------------------------------
template <bool A_IS_BF16, int OUT_MODE>
__global__ __launch_bounds__(256) void gemm_kernel(
    const void* __restrict__ Aop, const float* __restrict__ W,
    const float* __restrict__ bias,
    const float* __restrict__ CS, const float* __restrict__ SN,
    void* __restrict__ Out) {
  __shared__ bf16 As[128 * 40];   // A tile, row-major [m][k], padded
  __shared__ bf16 Bs[128 * 40];   // W tile transposed [n][k], padded

  const int tid  = threadIdx.x;
  const int lane = tid & 31;
  const int wid  = tid >> 5;
  const int wm   = wid & 3;   // 0..3
  const int wn   = wid >> 2;  // 0..1
  const int bm   = blockIdx.x * 128;
  const int bn   = blockIdx.y * 128;
  const int b    = blockIdx.z;

  const float* Af = (const float*)Aop + (size_t)b * Sc * DIMc;
  const bf16*  Ab = (const bf16*) Aop + (size_t)b * Sc * DIMc;

  v8f acc[2][4];
  const v8f vz = {0, 0, 0, 0, 0, 0, 0, 0};
#pragma unroll
  for (int i = 0; i < 2; ++i)
#pragma unroll
    for (int j = 0; j < 4; ++j) acc[i][j] = vz;

  const int arow = tid >> 1;          // 0..127
  const int akh  = (tid & 1) * 16;    // 0 or 16
  const int wkk  = tid >> 3;          // 0..31
  const int wnh  = (tid & 7) * 16;    // 0..112

  for (int k0 = 0; k0 < DIMc; k0 += 32) {
    // ---- stage A tile (convert f32 -> bf16 when needed) ----
    {
      __align__(16) bf16 tmp[16];
      if constexpr (A_IS_BF16) {
        const v8bf* src = (const v8bf*)(Ab + (size_t)(bm + arow) * DIMc + k0 + akh);
        *(v8bf*)&tmp[0] = src[0];
        *(v8bf*)&tmp[8] = src[1];
      } else {
        const float4* src = (const float4*)(Af + (size_t)(bm + arow) * DIMc + k0 + akh);
#pragma unroll
        for (int j = 0; j < 4; ++j) {
          float4 q = src[j];
          tmp[j * 4 + 0] = (bf16)q.x; tmp[j * 4 + 1] = (bf16)q.y;
          tmp[j * 4 + 2] = (bf16)q.z; tmp[j * 4 + 3] = (bf16)q.w;
        }
        if (k0 + 32 < DIMc)
          __builtin_prefetch(Af + (size_t)(bm + arow) * DIMc + k0 + 32 + akh, 0, 3);
      }
      v8bf* dst = (v8bf*)&As[arow * 40 + akh];
      dst[0] = *(v8bf*)&tmp[0];
      dst[1] = *(v8bf*)&tmp[8];
    }
    // ---- stage W tile, transposed into Bs[n][k] ----
    {
      const float4* src = (const float4*)(W + (size_t)(k0 + wkk) * DIMc + bn + wnh);
#pragma unroll
      for (int j = 0; j < 4; ++j) {
        float4 q = src[j];
        Bs[(wnh + j * 4 + 0) * 40 + wkk] = (bf16)q.x;
        Bs[(wnh + j * 4 + 1) * 40 + wkk] = (bf16)q.y;
        Bs[(wnh + j * 4 + 2) * 40 + wkk] = (bf16)q.z;
        Bs[(wnh + j * 4 + 3) * 40 + wkk] = (bf16)q.w;
      }
      if (k0 + 32 < DIMc)
        __builtin_prefetch(W + (size_t)(k0 + 32 + wkk) * DIMc + bn + wnh, 0, 3);
    }
    __syncthreads();

    v16bf af0 = load_afrag(&As[(wm * 32 + 0) * 40], 40, lane);
    v16bf af1 = load_afrag(&As[(wm * 32 + 16) * 40], 40, lane);
#pragma unroll
    for (int nt = 0; nt < 4; ++nt) {
      v16bf bfg = load_bfrag(&Bs[(wn * 64 + nt * 16) * 40], 40, lane);
      acc[0][nt] = wmma_bf16(af0, bfg, acc[0][nt]);
      acc[1][nt] = wmma_bf16(af1, bfg, acc[1][nt]);
    }
    __syncthreads();
  }

  // ---- epilogue: bias (+ fused RoPE) + store ----
  const int col0 = bn + wn * 64;
#pragma unroll
  for (int mt = 0; mt < 2; ++mt) {
#pragma unroll
    for (int nt = 0; nt < 4; ++nt) {
      const int n = col0 + nt * 16 + (lane & 15);
      const float bia = bias[n];
#pragma unroll
      for (int r = 0; r < 8; ++r) {
        const int s = bm + wm * 32 + mt * 16 + r + 8 * (lane >> 4);
        float v = acc[mt][nt][r] + bia;
        if constexpr (OUT_MODE == 2) {
          ((float*)Out)[((size_t)b * Sc + s) * DIMc + n] = v;
        } else {
          if constexpr (OUT_MODE == 0) {
            // RoPE: pairs (even,odd) live in adjacent lanes of the D-layout.
            const int i = (n & 63) >> 1;
            const float c  = CS[((size_t)b * Sc + s) * HALFc + i];
            const float sn = SN[((size_t)b * Sc + s) * HALFc + i];
            const float other = __shfl_xor(v, 1, 32);
            v = ((n & 1) == 0) ? (v * c - other * sn) : (other * sn + v * c);
          }
          const int h = n >> 6, dcol = n & 63;
          ((bf16*)Out)[(((size_t)b * Hc + h) * Sc + s) * Dc + dcol] = (bf16)v;
        }
      }
    }
  }
}

// ---------------------------------------------------------------------------
// Kernel 3: flash attention per (b,h). Block = 128 queries, 8 waves x 16 rows,
// streaming 128-key tiles with online softmax. QK^T and PV via WMMA; the V
// tile is DMA'd into LDS by the Tensor Data Mover when available.
// ---------------------------------------------------------------------------
__global__ __launch_bounds__(256) void attn_kernel(
    const bf16* __restrict__ Q, const bf16* __restrict__ Kk,
    const bf16* __restrict__ V, bf16* __restrict__ Ctx) {
  __shared__ bf16 VT[64 * 136];                     // V tile transposed [d][t]
  __shared__ bf16 Pl[8 * 16 * 136];                 // per-wave P tiles [16][128]
#if HAVE_TDM
  __shared__ __align__(128) bf16 Vrow[128 * 64];    // TDM destination [t][d]
#endif

  const int tid = threadIdx.x, lane = tid & 31, wid = tid >> 5;
  const int bh = blockIdx.y;           // b*H + h
  const int b = bh >> 4, h = bh & 15;
  const int q0 = blockIdx.x * 128;
  const size_t base = (size_t)bh * Sc * Dc;

  // Q fragments for this wave's 16 rows (K = d, 2 steps of 32)
  const int qrow = q0 + wid * 16;
  v16bf aq0 = load_afrag(Q + base + (size_t)qrow * Dc + 0, Dc, lane);
  v16bf aq1 = load_afrag(Q + base + (size_t)qrow * Dc + 32, Dc, lane);

  const v8f vz = {0, 0, 0, 0, 0, 0, 0, 0};
  v8f cacc[4];
#pragma unroll
  for (int i = 0; i < 4; ++i) cacc[i] = vz;
  float mrow[8], lrow[8];
#pragma unroll
  for (int r = 0; r < 8; ++r) { mrow[r] = -1e30f; lrow[r] = 0.0f; }

  bf16* Pw = &Pl[wid * 16 * 136];
  const int vkv = tid >> 1;            // 0..127
  const int vdh = (tid & 1) * 32;      // 0 or 32
#if HAVE_TDM
  const unsigned vrow_off = (unsigned)(uintptr_t)&Vrow[0];  // LDS byte offset
#endif

  for (int t0 = 0; t0 < Sc; t0 += 128) {
#if HAVE_TDM
    // ---- TDM: async DMA of the dense 128x64 bf16 V tile into LDS ----
    if (wid == 0) {
      tdm_load_2d_bf16(vrow_off, V + base + (size_t)t0 * Dc, Dc, 128);
      __builtin_amdgcn_s_wait_tensorcnt(0);
    }
    __syncthreads();
    // transpose LDS row-major tile -> VT[d][t]
    {
      __align__(16) bf16 tmp[32];
      const v8bf* src = (const v8bf*)&Vrow[vkv * Dc + vdh];
#pragma unroll
      for (int j = 0; j < 4; ++j) *(v8bf*)&tmp[j * 8] = src[j];
#pragma unroll
      for (int j = 0; j < 32; ++j) VT[(vdh + j) * 136 + vkv] = tmp[j];
    }
#else
    // ---- fallback: stage V tile transposed straight from global ----
    {
      __align__(16) bf16 tmp[32];
      const v8bf* src = (const v8bf*)(V + base + (size_t)(t0 + vkv) * Dc + vdh);
#pragma unroll
      for (int j = 0; j < 4; ++j) *(v8bf*)&tmp[j * 8] = src[j];
#pragma unroll
      for (int j = 0; j < 32; ++j) VT[(vdh + j) * 136 + vkv] = tmp[j];
    }
#endif
    __syncthreads();

    // ---- S = Q @ K^T : K fragments straight from global (k's [t][d]) ----
    v8f st[8];
    const bf16* Kbase = Kk + base + (size_t)t0 * Dc;
#pragma unroll
    for (int nt = 0; nt < 8; ++nt) {
      v16bf b0 = load_bfrag(Kbase + (size_t)(nt * 16) * Dc + 0, Dc, lane);
      v16bf b1 = load_bfrag(Kbase + (size_t)(nt * 16) * Dc + 32, Dc, lane);
      v8f z = wmma_bf16(aq0, b0, vz);
      st[nt] = wmma_bf16(aq1, b1, z);
    }

    // ---- online softmax per row (rows striped over 16-lane halves) ----
#pragma unroll
    for (int r = 0; r < 8; ++r) {
      float mx = -1e30f;
#pragma unroll
      for (int nt = 0; nt < 8; ++nt) mx = fmaxf(mx, st[nt][r]);
      mx = halfmax(mx) * 0.125f;                 // 1/sqrt(64)
      const float mnew = fmaxf(mrow[r], mx);
      const float corr = __expf(mrow[r] - mnew);
      mrow[r] = mnew;
      float rsum = 0.0f;
      const int prow = (r + 8 * (lane >> 4)) * 136 + (lane & 15);
#pragma unroll
      for (int nt = 0; nt < 8; ++nt) {
        const float p = __expf(st[nt][r] * 0.125f - mnew);
        rsum += p;
        Pw[prow + nt * 16] = (bf16)p;
      }
      rsum = halfsum(rsum);
      lrow[r] = lrow[r] * corr + rsum;
#pragma unroll
      for (int nt2 = 0; nt2 < 4; ++nt2) cacc[nt2][r] = cacc[nt2][r] * corr;
    }

    // ---- O += P @ V  (P from per-wave LDS, V^T shared) ----
#pragma unroll
    for (int ks = 0; ks < 4; ++ks) {
      v16bf ap = load_afrag(Pw + ks * 32, 136, lane);
#pragma unroll
      for (int nt2 = 0; nt2 < 4; ++nt2) {
        v16bf bv = load_bfrag(&VT[(nt2 * 16) * 136 + ks * 32], 136, lane);
        cacc[nt2] = wmma_bf16(ap, bv, cacc[nt2]);
      }
    }
    __syncthreads();
  }

  // ---- normalize and write context [B,S,DIM] bf16 ----
#pragma unroll
  for (int nt2 = 0; nt2 < 4; ++nt2) {
    const int dcol = nt2 * 16 + (lane & 15);
#pragma unroll
    for (int r = 0; r < 8; ++r) {
      const int s = q0 + wid * 16 + r + 8 * (lane >> 4);
      const float v = cacc[nt2][r] / lrow[r];
      Ctx[((size_t)b * Sc + s) * DIMc + h * Dc + dcol] = (bf16)v;
    }
  }
}

// ---------------------------------------------------------------------------
extern "C" void kernel_launch(void* const* d_in, const int* in_sizes, int n_in,
                              void* d_out, int out_size, void* d_ws, size_t ws_size,
                              hipStream_t stream) {
  (void)in_sizes; (void)n_in; (void)out_size; (void)ws_size;
  const float* x    = (const float*)d_in[0];
  const float* w2cs = (const float*)d_in[1];
  const float* Ksm  = (const float*)d_in[2];
  const float* Wq   = (const float*)d_in[3];
  const float* bq   = (const float*)d_in[4];
  const float* Wk   = (const float*)d_in[5];
  const float* bk   = (const float*)d_in[6];
  const float* Wv   = (const float*)d_in[7];
  const float* bv   = (const float*)d_in[8];
  const float* Wo   = (const float*)d_in[9];
  const float* bo   = (const float*)d_in[10];

  char* ws = (char*)d_ws;
  size_t off = 0;
  auto alloc = [&](size_t bytes) -> void* {
    void* p = ws + off;
    off += (bytes + 255) & ~(size_t)255;
    return p;
  };
  const size_t qkvBytes = (size_t)Bc * Hc * Sc * Dc * sizeof(bf16);
  bf16* qb  = (bf16*)alloc(qkvBytes);
  bf16* kb  = (bf16*)alloc(qkvBytes);
  bf16* vb  = (bf16*)alloc(qkvBytes);
  bf16* ctx = (bf16*)alloc((size_t)Bc * Sc * DIMc * sizeof(bf16));
  float* CS = (float*)alloc((size_t)Bc * Sc * HALFc * sizeof(float));
  float* SN = (float*)alloc((size_t)Bc * Sc * HALFc * sizeof(float));

  rope_kernel<<<dim3(Bc * Fc), 256, 0, stream>>>(w2cs, Ksm, CS, SN);

  dim3 gg(Sc / 128, DIMc / 128, Bc);
  gemm_kernel<false, 0><<<gg, 256, 0, stream>>>(x, Wq, bq, CS, SN, qb);
  gemm_kernel<false, 0><<<gg, 256, 0, stream>>>(x, Wk, bk, CS, SN, kb);
  gemm_kernel<false, 1><<<gg, 256, 0, stream>>>(x, Wv, bv, CS, SN, vb);

  attn_kernel<<<dim3(Sc / 128, Bc * Hc), 256, 0, stream>>>(qb, kb, vb, ctx);

  gemm_kernel<true, 2><<<gg, 256, 0, stream>>>(ctx, Wo, bo, CS, SN, d_out);
}
